// Binary_conv_bn_relu_58591943852107
// MI455X (gfx1250) — 8-bit matrix operand order repaired and checked by reading, compiled, not run on hardware
//
#include <hip/hip_runtime.h>
#include <hip/hip_bf16.h>

typedef int v8i __attribute__((ext_vector_type(8)));
typedef unsigned long long u64;

#define C_CH   256
#define H_DIM  56
#define W_DIM  56
#define HW     (H_DIM * W_DIM)
#define N_B    32
#define NHW    (N_B * HW)
#define HP     (H_DIM + 2)
#define WP     (W_DIM + 2)
#define MROWS  64
#define TILE_BYTES (MROWS * C_CH)

__global__ __launch_bounds__(256) void bn_stats_kernel(
    const float* __restrict__ x, const float* __restrict__ gamma,
    const float* __restrict__ beta, float* __restrict__ cA, float* __restrict__ cB) {
  int c = blockIdx.x;
  int tid = threadIdx.x;
  float s = 0.f, ss = 0.f;
  for (int idx = tid; idx < NHW; idx += 256) {
    int n = idx / HW;
    int i = idx - n * HW;
    float v = x[((size_t)n * C_CH + c) * HW + i];
    s += v;
    ss += v * v;
  }
  __shared__ float rs[256];
  __shared__ float rq[256];
  rs[tid] = s; rq[tid] = ss;
  __syncthreads();
  for (int off = 128; off > 0; off >>= 1) {
    if (tid < off) { rs[tid] += rs[tid + off]; rq[tid] += rq[tid + off]; }
    __syncthreads();
  }
  if (tid == 0) {
    const float inv_n = 1.0f / (float)NHW;
    float mean = rs[0] * inv_n;
    float var  = rq[0] * inv_n - mean * mean;
    float a = gamma[c] * rsqrtf(var + 1e-5f);
    cA[c] = a;
    cB[c] = beta[c] - mean * a;
  }
}

__global__ void zero_kernel(int4* __restrict__ p, size_t n4) {
  size_t i = (size_t)blockIdx.x * blockDim.x + threadIdx.x;
  if (i < n4) p[i] = make_int4(0, 0, 0, 0);
}

__global__ __launch_bounds__(256) void quantize_kernel(
    const float* __restrict__ x, const float* __restrict__ cA,
    const float* __restrict__ cB, char* __restrict__ xpad) {
  size_t idx = (size_t)blockIdx.x * blockDim.x + threadIdx.x;
  if (idx >= (size_t)N_B * C_CH * HW) return;
  int i   = (int)(idx % HW);
  int t   = (int)(idx / HW);
  int c   = t % C_CH;
  int n   = t / C_CH;
  int h   = i / W_DIM;
  int w   = i - h * W_DIM;
  float v = cA[c] * x[idx] + cB[c];
  char  q = v > 0.f ? (char)1 : (v < 0.f ? (char)-1 : (char)0);
  xpad[(((size_t)n * HP + (h + 1)) * WP + (w + 1)) * C_CH + c] = q;
}

__global__ __launch_bounds__(32) void pack_w_kernel(
    const float* __restrict__ w, char* __restrict__ wpack) {
  int bid  = blockIdx.x;
  int lane = threadIdx.x;
  int ks = bid >> 4, nt = bid & 15;
  int tap = ks >> 2, cb = ks & 3;
  int kh = tap / 3, kw = tap - kh * 3;
  int co = nt * 16 + (lane & 15);
  int* dst = (int*)wpack + ((size_t)bid * 32 + lane) * 8;
  for (int v = 0; v < 8; ++v) {
    int packed = 0;
    for (int byte = 0; byte < 4; ++byte) {
      int klocal = ((v >> 2) << 5) + ((lane >> 4) << 4) + ((v & 3) << 2) + byte;
      int ci = cb * 64 + klocal;
      float wv = w[(((size_t)co * C_CH + ci) * 3 + kh) * 3 + kw];
      int q = wv > 0.f ? 1 : (wv < 0.f ? -1 : 0);
      packed |= (q & 0xff) << (8 * byte);
    }
    dst[v] = packed;
  }
}

__global__ __launch_bounds__(512) void bconv_gemm_kernel(
    const char* __restrict__ xpad, const char* __restrict__ wpack,
    const float* __restrict__ bias, float* __restrict__ out) {
  __shared__ alignas(16) char tileA[2 * TILE_BYTES];

  const int mTile = blockIdx.x;
  const int tid   = threadIdx.x;
  const int wave  = tid >> 5;
  const int lane  = tid & 31;

  const unsigned ldsBase = (unsigned)(size_t)(void*)tileA;

  const int r     = tid >> 3;
  const int chunk = tid & 7;
  int m  = mTile * MROWS + r;
  int n_ = m / HW;
  int i_ = m - n_ * HW;
  int h  = i_ / W_DIM;
  int wc = i_ - h * W_DIM;
  const unsigned ldsRowOff = ldsBase + (unsigned)(r * 256 + chunk * 32);

  const int  frow = lane & 15;
  const int  half = lane >> 4;
  const unsigned aLane = (unsigned)(frow * 256 + half * 16);

  __builtin_prefetch(wpack + (size_t)wave * 32 * 32, 0, 3);

  v8i acc[4];
#pragma unroll
  for (int s = 0; s < 4; ++s) acc[s] = (v8i){0, 0, 0, 0, 0, 0, 0, 0};

  {
    u64 g0 = (u64)(xpad + ((((size_t)n_ * HP + h) * WP + wc) * C_CH + chunk * 32));
    unsigned l0 = ldsRowOff;
    asm volatile("global_load_async_to_lds_b128 %0, %1, off"
                 :: "v"(l0), "v"(g0) : "memory");
    asm volatile("global_load_async_to_lds_b128 %0, %1, off offset:16"
                 :: "v"(l0), "v"(g0) : "memory");
  }
  asm volatile("s_wait_asynccnt 0x0" ::: "memory");
  __syncthreads();

  for (int tap = 0; tap < 9; ++tap) {
    if (tap < 8) {
      int t1 = tap + 1;
      int kh = t1 / 3, kw = t1 - kh * 3;
      u64 g0 = (u64)(xpad +
                     ((((size_t)n_ * HP + (h + kh)) * WP + (wc + kw)) * C_CH + chunk * 32));
      unsigned l0 = (unsigned)(((tap + 1) & 1) * TILE_BYTES) + ldsRowOff;
      asm volatile("global_load_async_to_lds_b128 %0, %1, off"
                   :: "v"(l0), "v"(g0) : "memory");
      asm volatile("global_load_async_to_lds_b128 %0, %1, off offset:16"
                   :: "v"(l0), "v"(g0) : "memory");
    }

    const char* bufPtr = tileA + (tap & 1) * TILE_BYTES;
#pragma unroll
    for (int cb = 0; cb < 4; ++cb) {
      const int4* wq = (const int4*)wpack +
                       ((size_t)((tap * 4 + cb) * 16 + wave) * 32 + lane) * 2;
      int4 b0 = wq[0];
      int4 b1 = wq[1];
      v8i b;
      b[0] = b0.x; b[1] = b0.y; b[2] = b0.z; b[3] = b0.w;
      b[4] = b1.x; b[5] = b1.y; b[6] = b1.z; b[7] = b1.w;

#pragma unroll
      for (int sub = 0; sub < 4; ++sub) {
        const char* ap = bufPtr + sub * 16 * 256 + aLane + cb * 64;
        int2 c0 = *(const int2*)(ap);
        int2 c1 = *(const int2*)(ap + 8);
        int2 c2 = *(const int2*)(ap + 32);
        int2 c3 = *(const int2*)(ap + 40);
        v8i a;
        a[0] = c0.x; a[1] = c0.y; a[2] = c1.x; a[3] = c1.y;
        a[4] = c2.x; a[5] = c2.y; a[6] = c3.x; a[7] = c3.y;

        acc[sub] = __builtin_amdgcn_wmma_i32_16x16x64_iu8(true, a, true, b,
                                                          acc[sub], false, false);
      }
    }

    asm volatile("s_wait_asynccnt 0x0" ::: "memory");
    __syncthreads();
  }

  int co = wave * 16 + (lane & 15);
  float bs = bias[co];
  const int mBase = mTile * MROWS;
#pragma unroll
  for (int sub = 0; sub < 4; ++sub) {
#pragma unroll
    for (int j = 0; j < 8; ++j) {
      int mm = mBase + sub * 16 + j + half * 8;
      int nn = mm / HW;
      size_t oidx = (size_t)co * HW + (size_t)mm + (size_t)nn * ((C_CH - 1) * HW);
      float v = (float)acc[sub][j] + bs;
      v = v > 0.f ? v : 0.f;
      out[oidx] = v;
    }
  }
}

extern "C" void kernel_launch(void* const* d_in, const int* in_sizes, int n_in,
                              void* d_out, int out_size, void* d_ws, size_t ws_size,
                              hipStream_t stream) {
  const float* x      = (const float*)d_in[0];
  const float* gamma  = (const float*)d_in[1];
  const float* beta   = (const float*)d_in[2];
  const float* weight = (const float*)d_in[3];
  const float* bias   = (const float*)d_in[4];
  float* out = (float*)d_out;

  char* ws = (char*)d_ws;
  float* cA   = (float*)ws;
  float* cB   = cA + 256;
  char* wpack = ws + 4096;
  char* xpad  = ws + (1 << 20);

  bn_stats_kernel<<<C_CH, 256, 0, stream>>>(x, gamma, beta, cA, cB);

  size_t padBytes = (size_t)N_B * HP * WP * C_CH;
  size_t n4 = padBytes / 16;
  zero_kernel<<<(unsigned)((n4 + 255) / 256), 256, 0, stream>>>((int4*)xpad, n4);

  size_t nel = (size_t)N_B * C_CH * HW;
  quantize_kernel<<<(unsigned)((nel + 255) / 256), 256, 0, stream>>>(x, cA, cB, xpad);

  pack_w_kernel<<<576, 32, 0, stream>>>(weight, wpack);

  bconv_gemm_kernel<<<NHW / MROWS, 512, 0, stream>>>(xpad, wpack, bias, out);
}
